// Generator_33741263077950
// MI455X (gfx1250) — matrix operand order repaired and checked by reading, not compiled, not run on hardware
//
#include <hip/hip_runtime.h>
#include <math.h>

// ---------------------------------------------------------------------------
// StyleGAN2-like generator for MI455X (gfx1250, wave32).
// Convs = implicit GEMM with v_wmma_f32_16x16x32_f16 (f32 accum).
// K dimension is tap-major (k = tap*512 + i) and activations are staged in
// NHWC f16, so both A and B tiles stream through LDS as pure b128 traffic
// with loop-invariant tap decode (no per-element div/mod or predication).
// ---------------------------------------------------------------------------

typedef __attribute__((ext_vector_type(16))) _Float16 v16h;
typedef __attribute__((ext_vector_type(8)))  _Float16 v8h;
typedef __attribute__((ext_vector_type(8)))  float    v8f;

#define SQRT2F 1.41421356237309504880f
#define CCH   512
#define KTOT  4608            // 9 taps * 512 channels
#define RSQ512 0.044194173824159216f   // 1/sqrt(512)

// ------------------------- mapping network --------------------------------
__global__ void k_mapping(const float* __restrict__ styles,
                          const float* __restrict__ w0, const float* __restrict__ b0,
                          const float* __restrict__ w1, const float* __restrict__ b1,
                          float* __restrict__ wmap) {
  __shared__ float xb[512];
  __shared__ float yb[512];
  __shared__ float red[512];
  const int b = blockIdx.x, t = threadIdx.x;
  float v = styles[b * 512 + t];
  red[t] = v * v;
  __syncthreads();
  for (int s = 256; s > 0; s >>= 1) {
    if (t < s) red[t] += red[t + s];
    __syncthreads();
  }
  const float inv = rsqrtf(red[0] * (1.0f / 512.0f) + 1e-8f);
  xb[t] = v * inv;
  __syncthreads();
  const float scale = RSQ512 * 0.01f;  // (1/sqrt(512)) * lr_mul
  float acc = 0.f;
  for (int i = 0; i < 512; ++i) acc += xb[i] * w0[t * 512 + i];
  acc = acc * scale + b0[t] * 0.01f;
  acc = (acc > 0.f ? acc : 0.2f * acc) * SQRT2F;
  yb[t] = acc;
  __syncthreads();
  float acc2 = 0.f;
  for (int i = 0; i < 512; ++i) acc2 += yb[i] * w1[t * 512 + i];
  acc2 = acc2 * scale + b1[t] * 0.01f;
  acc2 = (acc2 > 0.f ? acc2 : 0.2f * acc2) * SQRT2F;
  wmap[b * 512 + t] = acc2;
}

// ------------------- per-layer style vector s[b,i] ------------------------
__global__ void k_style(const float* __restrict__ wmap, const float* __restrict__ mod_w,
                        const float* __restrict__ mod_b, float* __restrict__ s) {
  __shared__ float wrow[512];
  const int b = blockIdx.y;
  const int i = blockIdx.x * 256 + threadIdx.x;
  for (int j = threadIdx.x; j < 512; j += 256) wrow[j] = wmap[b * 512 + j];
  __syncthreads();
  float acc = 0.f;
  for (int j = 0; j < 512; ++j) acc += wrow[j] * mod_w[i * 512 + j];
  s[b * 512 + i] = acc * RSQ512 + mod_b[i];
}

// -- weight prep: f32 -> f16 A-matrix [O][tap][512] (+opt flip) and WW -----
__global__ void k_prep(const float* __restrict__ w, _Float16* __restrict__ A,
                       float* __restrict__ WW, int flip) {
  const int i = blockIdx.x * 256 + threadIdx.x;   // in-channel
  const int o = blockIdx.y;                       // out-channel
  const float cs = 0.014731391274719742f;         // 1/sqrt(512*9)
  const float* wp = w + ((size_t)o * 512 + i) * 9;
  float tv[9];
  float sum = 0.f;
  for (int t = 0; t < 9; ++t) { float v = wp[t] * cs; tv[t] = v; sum += v * v; }
  WW[o * 512 + i] = sum;
  _Float16* ap = A + (size_t)o * KTOT + i;        // tap-major K: k = t*512 + i
  for (int t = 0; t < 9; ++t) ap[t * 512] = (_Float16)(flip ? tv[8 - t] : tv[t]);
}

// --------------------- demod coefficients d[b,o] --------------------------
__global__ void k_demod(const float* __restrict__ WW, const float* __restrict__ s,
                        float* __restrict__ dv) {
  __shared__ float s2[512];
  const int b = blockIdx.y;
  const int o = blockIdx.x * 256 + threadIdx.x;
  for (int j = threadIdx.x; j < 512; j += 256) { float v = s[b * 512 + j]; s2[j] = v * v; }
  __syncthreads();
  float acc = 0.f;
  for (int i = 0; i < 512; ++i) acc += WW[o * 512 + i] * s2[i];
  dv[b * 512 + o] = rsqrtf(acc + 1e-8f);
}

// ---- premodulate + NCHW->NHWC: xs[((b*H+y)*H+x)*512+i] = f16(x * s) ------
__global__ void k_premod(const _Float16* __restrict__ x, const float* __restrict__ cst,
                         const float* __restrict__ s, _Float16* __restrict__ xout,
                         int HW, int fromConst, int total) {
  const int idx = blockIdx.x * 256 + threadIdx.x;
  if (idx >= total) return;
  const int i  = idx & 511;
  const int p  = idx >> 9;          // pixel index (b,y,x)
  const int hw = p % HW;
  const int b  = p / HW;
  const float v = fromConst ? cst[i * HW + hw]
                            : (float)x[((size_t)b * CCH + i) * HW + hw];
  xout[idx] = (_Float16)(v * s[b * CCH + i]);
}

// ----------------- WMMA implicit-GEMM convolution -------------------------
// WG: 256 threads (8 waves) -> 128(M out-ch) x 128(N spatial) tile.
// Wave: 32(M) x 64(N) = 2x4 WMMA tiles, 8 accumulators.
// mode 0: 3x3 pad-1 conv (Hout==Hin), epilogue = demod+bias+lrelu*sqrt2.
// mode 1: stride-2 transposed 3x3 (Hout==2*Hin+1, flipped A), demod only.
__global__ __launch_bounds__(256)
void k_conv(const _Float16* __restrict__ xs,   // NHWC [B][Hin][Hin][512]
            const _Float16* __restrict__ A,    // [512][9*512]
            const float* __restrict__ dvec, const float* __restrict__ abias,
            _Float16* __restrict__ out,        // NCHW f16 [B][512][Hout][Hout]
            int Hin, int Hout, int mode) {
  __shared__ _Float16 lA[128 * 40];  // [128 M][32 K], stride 40 halfs (16B-aligned rows)
  __shared__ _Float16 lB[128 * 40];  // [128 N][32 K]
  const int b      = blockIdx.z;
  const int blockM = blockIdx.y * 128;
  const int nBase  = blockIdx.x * 128;
  const int Nsp    = Hout * Hout;
  const int tid  = threadIdx.x;
  const int lane = tid & 31;
  const int wv   = tid >> 5;
  const int mBlk = wv & 3;           // 4 M sub-blocks of 32
  const int nBlk = wv >> 2;          // 2 N sub-blocks of 64
  const int lm   = lane & 15;
  const int kbA  = (lane & 16) ? 8  : 0;
  const int kbB  = (lane & 16) ? 8  : 0;

  v8f acc[2][4] = {};

  // cooperative staging assignment: thread -> (row, 16-wide K chunk)
  const int rS = tid >> 1;           // 0..127
  const int ch = tid & 1;            // 0..1
  int yG = 0, xG = 0; bool nValid = false;
  { const int nsp = nBase + rS;
    if (nsp < Nsp) { nValid = true; yG = nsp / Hout; xG = nsp - yG * Hout; } }

  const _Float16* gA = A + (size_t)(blockM + rS) * KTOT + ch * 16;

  for (int t = 0; t < 9; ++t) {
    const int ky = t / 3;
    const int kx = t - ky * 3;
    // loop-invariant tap decode: per-thread source pixel pointer + validity
    const _Float16* src = xs;
    bool ok = false;
    if (nValid) {
      if (mode == 0) {
        const int iy = yG + ky - 1, ix = xG + kx - 1;
        if (iy >= 0 && iy < Hin && ix >= 0 && ix < Hin) {
          ok = true;
          src = xs + (((size_t)b * Hin + iy) * Hin + ix) * CCH;
        }
      } else {
        const int py = yG + ky - 2, px = xG + kx - 2;
        if (py >= 0 && px >= 0 && ((py | px) & 1) == 0) {
          const int iy = py >> 1, ix = px >> 1;
          if (iy < Hin && ix < Hin) {
            ok = true;
            src = xs + (((size_t)b * Hin + iy) * Hin + ix) * CCH;
          }
        }
      }
    }
    for (int kc = 0; kc < 16; ++kc) {
      const int k0 = t * 512 + kc * 32;
      // ---- stage A tile: 2 x global b128 -> 2 x ds b128 per thread ----
      {
        const _Float16* gp = gA + k0;
        if (kc + 1 < 16) __builtin_prefetch(gp + 32, 0, 1);
        v8h a0 = *(const v8h*)(gp);
        v8h a1 = *(const v8h*)(gp + 8);
        *(v8h*)&lA[rS * 40 + ch * 16]     = a0;
        *(v8h*)&lA[rS * 40 + ch * 16 + 8] = a1;
      }
      // ---- stage B tile: 16 contiguous NHWC channels ----
      {
        const int ib = kc * 32 + ch * 16;
        v8h b0 = {}; v8h b1 = {};
        if (ok) { b0 = *(const v8h*)(src + ib); b1 = *(const v8h*)(src + ib + 8); }
        *(v8h*)&lB[rS * 40 + ch * 16]     = b0;
        *(v8h*)&lB[rS * 40 + ch * 16 + 8] = b1;
      }
      __syncthreads();
      // ---- fragments (b128 LDS) + 8 WMMAs per wave ----
      v16h av[2];
      #pragma unroll
      for (int mt = 0; mt < 2; ++mt) {
        const int mr = mBlk * 32 + mt * 16 + lm;
        v8h a0 = *(const v8h*)&lA[mr * 40 + kbA];
        v8h a1 = *(const v8h*)&lA[mr * 40 + kbA + 16];
        av[mt] = __builtin_shufflevector(a0, a1,
                   0,1,2,3,4,5,6,7,8,9,10,11,12,13,14,15);
      }
      #pragma unroll
      for (int nt = 0; nt < 4; ++nt) {
        const int nr = nBlk * 64 + nt * 16 + lm;
        v8h b0 = *(const v8h*)&lB[nr * 40 + kbB];
        v8h b1 = *(const v8h*)&lB[nr * 40 + kbB + 16];
        v16h bv = __builtin_shufflevector(b0, b1,
                    0,1,2,3,4,5,6,7,8,9,10,11,12,13,14,15);
        acc[0][nt] = __builtin_amdgcn_wmma_f32_16x16x32_f16(false, av[0], false, bv,
                                                            (short)0, acc[0][nt], false, false);
        acc[1][nt] = __builtin_amdgcn_wmma_f32_16x16x32_f16(false, av[1], false, bv,
                                                            (short)0, acc[1][nt], false, false);
      }
      __syncthreads();
    }
  }
  // ---- epilogue: C/D layout lane L, vgpr r -> M = r + 8*(L>=16), N = L%16
  #pragma unroll
  for (int mt = 0; mt < 2; ++mt) {
    const int oBase = blockM + mBlk * 32 + mt * 16 + (lane >> 4) * 8;
    float dv[8], ab[8];
    #pragma unroll
    for (int r = 0; r < 8; ++r) {
      dv[r] = dvec[b * CCH + oBase + r];
      ab[r] = (mode == 0) ? abias[oBase + r] : 0.f;
    }
    #pragma unroll
    for (int nt = 0; nt < 4; ++nt) {
      const int n = nBase + nBlk * 64 + nt * 16 + lm;
      if (n < Nsp) {
        #pragma unroll
        for (int r = 0; r < 8; ++r) {
          float v = acc[mt][nt][r] * dv[r];
          if (mode == 0) { v += ab[r]; v = (v > 0.f ? v : 0.2f * v) * SQRT2F; }
          out[((size_t)b * CCH + oBase + r) * Nsp + n] = (_Float16)v;
        }
      }
    }
  }
}

// -------- blur after transposed conv (+ bias + lrelu*sqrt2) ---------------
__global__ void k_blur(const _Float16* __restrict__ tin, const float* __restrict__ abias,
                       _Float16* __restrict__ xout, int Ht, int H2, int total) {
  const int idx = blockIdx.x * 256 + threadIdx.x;
  if (idx >= total) return;
  const int x = idx % H2;
  int t1 = idx / H2;
  const int y = t1 % H2;  t1 /= H2;
  const int o = t1 % CCH;
  const int b = t1 / CCH;
  const float kk[4] = {0.25f, 0.75f, 0.75f, 0.25f};   // outer product = BLUR_K*4
  float acc = 0.f;
  const _Float16* base = tin + ((size_t)b * CCH + o) * Ht * Ht;
  for (int ky = 0; ky < 4; ++ky) {
    const int iy = y + ky - 1;
    if (iy < 0 || iy >= Ht) continue;
    for (int kx = 0; kx < 4; ++kx) {
      const int ix = x + kx - 1;
      if (ix < 0 || ix >= Ht) continue;
      acc += kk[ky] * kk[kx] * (float)base[iy * Ht + ix];
    }
  }
  acc += abias[o];
  acc = (acc > 0.f ? acc : 0.2f * acc) * SQRT2F;
  xout[idx] = (_Float16)acc;
}

// ------------------------- to_rgb (1x1, O=3) ------------------------------
__global__ void k_rgb(const _Float16* __restrict__ x, const float* __restrict__ w,
                      const float* __restrict__ s, const float* __restrict__ bias,
                      float* __restrict__ outp, int HW) {
  const int p = blockIdx.x * 256 + threadIdx.x;
  const int b = blockIdx.y;
  if (p >= HW) return;
  float a0 = 0.f, a1 = 0.f, a2 = 0.f;
  for (int i = 0; i < 512; ++i) {
    const float xm = (float)x[((size_t)b * CCH + i) * HW + p] * s[b * 512 + i];
    a0 += w[i] * xm;
    a1 += w[512 + i] * xm;
    a2 += w[1024 + i] * xm;
  }
  outp[((size_t)b * 3 + 0) * HW + p] = a0 * RSQ512 + bias[0];
  outp[((size_t)b * 3 + 1) * HW + p] = a1 * RSQ512 + bias[1];
  outp[((size_t)b * 3 + 2) * HW + p] = a2 * RSQ512 + bias[2];
}

// -------------- skip pyramid: out = rgb + upfirdn(skip,up2) ---------------
__global__ void k_skip(const float* __restrict__ rgb, const float* __restrict__ sp,
                       float* __restrict__ outp, int Hin, int total) {
  const int idx = blockIdx.x * 256 + threadIdx.x;
  if (idx >= total) return;
  const int H2 = Hin * 2;
  const int x = idx % H2;
  int t1 = idx / H2;
  const int y = t1 % H2;  t1 /= H2;     // t1 = b*3 + c
  const float kk[4] = {0.25f, 0.75f, 0.75f, 0.25f};
  float acc = rgb[idx];
  const float* base = sp + (size_t)t1 * Hin * Hin;
  for (int ky = (y & 1); ky < 4; ky += 2) {
    const int iy = (y + ky - 2) >> 1;
    if (iy < 0 || iy >= Hin) continue;
    for (int kx = (x & 1); kx < 4; kx += 2) {
      const int ix = (x + kx - 2) >> 1;
      if (ix < 0 || ix >= Hin) continue;
      acc += kk[ky] * kk[kx] * base[iy * Hin + ix];
    }
  }
  outp[idx] = acc;
}

// ---------------------------------------------------------------------------
extern "C" void kernel_launch(void* const* d_in, const int* in_sizes, int n_in,
                              void* d_out, int out_size, void* d_ws, size_t ws_size,
                              hipStream_t stream) {
  (void)in_sizes; (void)n_in; (void)out_size; (void)ws_size;
  // pytree order (dict-insertion, recursive): styles, style[0]{w,b}, style[1]{w,b},
  // const, conv1{weight,mod_w,mod_b,act_bias}, to_rgb1{weight,mod_w,mod_b,bias},
  // convs[0..7]{...}, to_rgbs[0..3]{...}
  const float* styles = (const float*)d_in[0];
  const float* sw0 = (const float*)d_in[1];
  const float* sb0 = (const float*)d_in[2];
  const float* sw1 = (const float*)d_in[3];
  const float* sb1 = (const float*)d_in[4];
  const float* cst = (const float*)d_in[5];
  const float *cw[9], *cmw[9], *cmb[9], *cab[9];
  cw[0] = (const float*)d_in[6];  cmw[0] = (const float*)d_in[7];
  cmb[0] = (const float*)d_in[8]; cab[0] = (const float*)d_in[9];
  for (int j = 0; j < 8; ++j) {
    const int base = 14 + 4 * j;
    cw[j + 1]  = (const float*)d_in[base + 0];
    cmw[j + 1] = (const float*)d_in[base + 1];
    cmb[j + 1] = (const float*)d_in[base + 2];
    cab[j + 1] = (const float*)d_in[base + 3];
  }
  const float *rw[5], *rmw[5], *rmb[5], *rb[5];
  rw[0] = (const float*)d_in[10]; rmw[0] = (const float*)d_in[11];
  rmb[0] = (const float*)d_in[12]; rb[0] = (const float*)d_in[13];
  for (int j = 0; j < 4; ++j) {
    const int base = 46 + 4 * j;
    rw[j + 1]  = (const float*)d_in[base + 0];
    rmw[j + 1] = (const float*)d_in[base + 1];
    rmb[j + 1] = (const float*)d_in[base + 2];
    rb[j + 1]  = (const float*)d_in[base + 3];
  }

  // ---------------- workspace carve (256B aligned) ----------------
  char* ws = (char*)d_ws;
  size_t off = 0;
  auto carve = [&](size_t bytes) -> char* {
    char* p = ws + off;
    off += (bytes + 255) & ~(size_t)255;
    return p;
  };
  float* wmap = (float*)carve((size_t)16 * 512 * 4);
  float* sconv[9]; for (int l = 0; l < 9; ++l) sconv[l] = (float*)carve((size_t)16 * 512 * 4);
  float* srgb[5];  for (int l = 0; l < 5; ++l) srgb[l]  = (float*)carve((size_t)16 * 512 * 4);
  float* dconv[9]; for (int l = 0; l < 9; ++l) dconv[l] = (float*)carve((size_t)16 * 512 * 4);
  float* WW = (float*)carve((size_t)512 * 512 * 4);
  _Float16* Amat[9];
  for (int l = 0; l < 9; ++l) Amat[l] = (_Float16*)carve((size_t)512 * KTOT * 2);
  const size_t actB = (size_t)16 * 512 * 64 * 64 * 2;
  _Float16* xA  = (_Float16*)carve(actB);            // NCHW f16 activations
  _Float16* xB  = (_Float16*)carve(actB);
  _Float16* xsb = (_Float16*)carve(actB);            // NHWC f16 premodulated
  _Float16* tmp = (_Float16*)carve((size_t)16 * 512 * 65 * 65 * 2);
  float* rgbT  = (float*)carve((size_t)16 * 3 * 64 * 64 * 4);
  float* skip0 = (float*)carve((size_t)16 * 3 * 64 * 64 * 4);
  float* skip1 = (float*)carve((size_t)16 * 3 * 64 * 64 * 4);

  // ---------------- per-launch precompute ----------------
  k_mapping<<<16, 512, 0, stream>>>(styles, sw0, sb0, sw1, sb1, wmap);
  const dim3 gs(2, 16);
  for (int l = 0; l < 9; ++l) {
    const int flip = (l == 1 || l == 3 || l == 5 || l == 7) ? 1 : 0; // upsample layers
    k_style<<<gs, 256, 0, stream>>>(wmap, cmw[l], cmb[l], sconv[l]);
    k_prep<<<dim3(2, 512), 256, 0, stream>>>(cw[l], Amat[l], WW, flip);
    k_demod<<<gs, 256, 0, stream>>>(WW, sconv[l], dconv[l]);
  }
  for (int l = 0; l < 5; ++l)
    k_style<<<gs, 256, 0, stream>>>(wmap, rmw[l], rmb[l], srgb[l]);

  auto premod = [&](const _Float16* xin, const float* s, _Float16* xo, int H, int fromC) {
    const int total = 16 * 512 * H * H;
    k_premod<<<(total + 255) / 256, 256, 0, stream>>>(xin, cst, s, xo, H * H, fromC, total);
  };
  auto conv = [&](const _Float16* xsrc, int l, _Float16* outp, int Hin, int Hout, int mode) {
    const int Nsp = Hout * Hout;
    dim3 g((Nsp + 127) / 128, 4, 16);
    k_conv<<<g, 256, 0, stream>>>(xsrc, Amat[l], dconv[l], cab[l], outp, Hin, Hout, mode);
  };

  // ---------------- synthesis ----------------
  premod(nullptr, sconv[0], xsb, 4, 1);      // modulated const input (NHWC)
  conv(xsb, 0, xA, 4, 4, 0);                 // conv1 @ 4x4
  k_rgb<<<dim3(1, 16), 256, 0, stream>>>(xA, rw[0], srgb[0], rb[0], skip0, 16);

  float* skipCur = skip0;
  float* skipNext = skip1;
  for (int i = 0; i < 4; ++i) {
    const int Hin = 4 << i, Ht = 2 * Hin + 1, H2 = 2 * Hin;
    const int lu = 1 + 2 * i, ln = 2 + 2 * i, rl = 1 + i;
    // upsample styled_conv
    premod(xA, sconv[lu], xsb, Hin, 0);
    conv(xsb, lu, tmp, Hin, Ht, 1);
    {
      const int total = 16 * 512 * H2 * H2;
      k_blur<<<(total + 255) / 256, 256, 0, stream>>>(tmp, cab[lu], xB, Ht, H2, total);
    }
    // plain styled_conv
    premod(xB, sconv[ln], xsb, H2, 0);
    conv(xsb, ln, xA, H2, H2, 0);
    // to_rgb + skip pyramid
    {
      const int HW = H2 * H2;
      k_rgb<<<dim3((HW + 255) / 256, 16), 256, 0, stream>>>(xA, rw[rl], srgb[rl], rb[rl], rgbT, HW);
    }
    {
      const int total = 16 * 3 * H2 * H2;
      float* dst = (i == 3) ? (float*)d_out : skipNext;
      k_skip<<<(total + 255) / 256, 256, 0, stream>>>(rgbT, skipCur, dst, Hin, total);
      float* t = skipCur; skipCur = skipNext; skipNext = t;
    }
  }
}
